// LOFPostProcessor_89026082111909
// MI455X (gfx1250) — 8-bit matrix operand order repaired and checked by reading, compiled, not run on hardware
//
#include <hip/hip_runtime.h>
#include <stdint.h>

#define NIMG   8
#define NCLS   80
#define HH     200
#define WW     200
#define PPIX   (HH*WW)
#define TILE   32
#define KCAND  1024
#define TOPK   1000
#define OUTK   100
#define NEG_INF_F (-1e30f)
#define LOGIT_TH  (-2.9444389791664403f)
#define IOU_TH    0.6f

typedef unsigned int  u32;
typedef unsigned short u16;
typedef u32 tdm_u32x4 __attribute__((ext_vector_type(4)));
typedef int tdm_i32x8 __attribute__((ext_vector_type(8)));
typedef int tdm_i32x4 __attribute__((ext_vector_type(4)));
typedef int wmma_i32x8 __attribute__((ext_vector_type(8)));

__device__ __forceinline__ u32 fkey(float f) {
  u32 u = __float_as_uint(f);
  return (u & 0x80000000u) ? ~u : (u | 0x80000000u);
}
__device__ __forceinline__ float sigmoidf(float x) { return 1.0f / (1.0f + __expf(-x)); }
__device__ __forceinline__ u32 ballot32(bool p) {
#if defined(__AMDGCN__) && __has_builtin(__builtin_amdgcn_ballot_w32)
  return __builtin_amdgcn_ballot_w32(p);
#else
  return (u32)__ballot(p);
#endif
}

__global__ void __launch_bounds__(TILE) lof_score_kernel(
    const float* __restrict__ box_cls, const float* __restrict__ centerness,
    float* __restrict__ out_score, int* __restrict__ out_cls)
{
  __shared__ float tile[NCLS * TILE];
  const int n  = blockIdx.y;
  const int p0 = blockIdx.x * TILE;
  const int t  = threadIdx.x;
  const float* gbase = box_cls + (size_t)n * NCLS * PPIX + p0;

#if defined(__AMDGCN__) && __has_builtin(__builtin_amdgcn_tensor_load_to_lds)
  {
    unsigned long long ga = (unsigned long long)(uintptr_t)gbase;
    u32 lds = (u32)(uintptr_t)(void*)tile;
    tdm_u32x4 g0;
    g0[0] = 1u;
    g0[1] = lds;
    g0[2] = (u32)ga;
    g0[3] = ((u32)(ga >> 32) & 0x01FFFFFFu) | (2u << 30);
    tdm_i32x8 g1;
    const u32 td0 = (u32)PPIX, td1 = (u32)NCLS;
    const unsigned long long s0 = (unsigned long long)PPIX;
    g1[0] = (int)(2u << 16);
    g1[1] = (int)((td0 & 0xFFFFu) << 16);
    g1[2] = (int)((td0 >> 16) | ((td1 & 0xFFFFu) << 16));
    g1[3] = (int)((td1 >> 16) | ((u32)TILE << 16));
    g1[4] = (int)(u32)NCLS;
    g1[5] = (int)(u32)(s0 & 0xFFFFFFFFull);
    g1[6] = (int)(u32)((s0 >> 32) & 0xFFFFull);
    g1[7] = 0;
    tdm_i32x4 gz = (tdm_i32x4)0;
#if __has_include(<hip/amd_detail/amd_gfx1250_TDM.h>)
    tdm_i32x8 gz8 = (tdm_i32x8)0;
    __builtin_amdgcn_tensor_load_to_lds(g0, g1, gz, gz, gz8, 0);
#else
    __builtin_amdgcn_tensor_load_to_lds(g0, g1, gz, gz, 0);
#endif
#if __has_builtin(__builtin_amdgcn_s_wait_tensorcnt)
    __builtin_amdgcn_s_wait_tensorcnt(0);
#else
    asm volatile("s_wait_tensorcnt 0x0" ::: "memory");
#endif
    asm volatile("" :: "v"(&tile[0]) : "memory");
  }
#else
  for (int c = 0; c < NCLS; ++c) tile[c * TILE + t] = gbase[(size_t)c * PPIX + t];
#endif
  __syncthreads();

  const int p = p0 + t;
  float m = tile[t]; int am = 0;
#pragma unroll 8
  for (int c = 1; c < NCLS; ++c) {
    float v = tile[c * TILE + t];
    if (v > m) { m = v; am = c; }
  }
  const float ctr = centerness[(size_t)n * PPIX + p];
  const float s = sigmoidf(m) * sigmoidf(ctr);
  out_score[(size_t)n * PPIX + p] = (m > LOGIT_TH) ? s : NEG_INF_F;
  out_cls[(size_t)n * PPIX + p] = am + 1;
}

__global__ void __launch_bounds__(1024) lof_topk_kernel(
    const float* __restrict__ scores, const int* __restrict__ cls,
    const float* __restrict__ loc, const float* __restrict__ reg,
    const float* __restrict__ lof, const int* __restrict__ imsz,
    float* __restrict__ o_score, float* __restrict__ o_boxes,
    int* __restrict__ o_label, int* __restrict__ o_tag, u32* __restrict__ o_vbits)
{
  __shared__ u32 hist[256];
  __shared__ u32 sh_sel, sh_need;
  __shared__ float s_sc[KCAND];
  __shared__ int   s_ix[KCAND];
  __shared__ u32 s_cg, s_ce;
  const int n = blockIdx.x;
  const int t = threadIdx.x;
  const float* sc = scores + (size_t)n * PPIX;

  u32 prefix = 0, pmask = 0, need = TOPK;
  for (int pass = 0; pass < 4; ++pass) {
    const int shift = 24 - 8 * pass;
    if (t < 256) hist[t] = 0;
    __syncthreads();
    for (int e = t; e < PPIX; e += 1024) {
      u32 u = fkey(sc[e]);
      if ((u & pmask) == prefix) atomicAdd(&hist[(u >> shift) & 255u], 1u);
    }
    __syncthreads();
    if (t == 0) {
      u32 cum = 0; int b = 255;
      for (; b > 0; --b) { u32 c0 = hist[b]; if (cum + c0 >= need) break; cum += c0; }
      sh_sel = (u32)b; sh_need = need - cum;
    }
    __syncthreads();
    prefix |= sh_sel << shift; pmask |= 0xFFu << shift; need = sh_need;
    __syncthreads();
  }
  const u32 uth = prefix;
  const u32 needEq = need;
  const u32 nGreat = TOPK - needEq;

  s_sc[t] = -3.0e38f; s_ix[t] = 0;
  if (t == 0) { s_cg = 0; s_ce = 0; }
  __syncthreads();
  for (int e = t; e < PPIX; e += 1024) {
    float v = sc[e]; u32 u = fkey(v);
    if (u > uth) {
      u32 pos = atomicAdd(&s_cg, 1u);
      if (pos < KCAND) { s_sc[pos] = v; s_ix[pos] = e; }
    } else if (u == uth) {
      u32 e2 = atomicAdd(&s_ce, 1u);
      if (e2 < needEq) { u32 pos = nGreat + e2; s_sc[pos] = v; s_ix[pos] = e; }
    }
  }
  __syncthreads();

  for (u32 k = 2; k <= KCAND; k <<= 1) {
    for (u32 j = k >> 1; j > 0; j >>= 1) {
      __syncthreads();
      u32 ixj = (u32)t ^ j;
      if (ixj > (u32)t) {
        float sa = s_sc[t], sb = s_sc[ixj];
        int ia = s_ix[t], ib = s_ix[ixj];
        u32 ka = fkey(sa), kb = fkey(sb);
        bool aFirst = (ka > kb) || (ka == kb && ia < ib);
        bool desc = (((u32)t & k) == 0);
        if (desc ? !aFirst : aFirst) {
          s_sc[t] = sb; s_sc[ixj] = sa; s_ix[t] = ib; s_ix[ixj] = ia;
        }
      }
    }
  }
  __syncthreads();

  const float wmax = (float)(imsz[n * 2 + 1] - 1);
  const float hmax = (float)(imsz[n * 2 + 0] - 1);
  float scv = s_sc[t]; int p = s_ix[t];
  bool val = scv > (NEG_INF_F * 0.5f);
  float x1 = 0.f, y1 = 0.f, x2 = 0.f, y2 = 0.f; int lb = 0, tg = 0;
  if (val) {
    float lx = loc[2 * p], ly = loc[2 * p + 1];
    const size_t rb = (size_t)n * 4 * PPIX + p;
    float rl = reg[rb], rt = reg[rb + PPIX], rr = reg[rb + 2 * PPIX], rbm = reg[rb + 3 * PPIX];
    x1 = fminf(fmaxf(lx - rl,  0.f), wmax);
    y1 = fminf(fmaxf(ly - rt,  0.f), hmax);
    x2 = fminf(fmaxf(lx + rr,  0.f), wmax);
    y2 = fminf(fmaxf(ly + rbm, 0.f), hmax);
    lb = cls[(size_t)n * PPIX + p];
    tg = (int)rintf(lof[(size_t)n * PPIX + p]);
  }
  const int o = n * KCAND + t;
  o_score[o] = scv;
  o_boxes[o * 4 + 0] = x1; o_boxes[o * 4 + 1] = y1;
  o_boxes[o * 4 + 2] = x2; o_boxes[o * 4 + 3] = y2;
  o_label[o] = lb; o_tag[o] = tg;
  u32 bal = ballot32(val);
  if ((t & 31) == 0) o_vbits[n * 32 + (t >> 5)] = bal;
}

__global__ void __launch_bounds__(32) lof_nms_mask_kernel(
    const float* __restrict__ boxes, const int* __restrict__ labels,
    u16* __restrict__ mask)
{
  const int ti = blockIdx.x, tj = blockIdx.y, n = blockIdx.z;
  const int l = threadIdx.x;
  const int q = l & 15, hi = l >> 4;
  u16* mbase = mask + (size_t)n * KCAND * 64;
  if (tj < ti) {
    if (l < 16) mbase[(size_t)(ti * 16 + l) * 64 + tj] = 0;
    return;
  }
  const int* labn = labels + n * KCAND;
  const float* bx = boxes + (size_t)n * KCAND * 4;
  const int la  = labn[ti * 16 + q] - 1;
  const int lbv = labn[tj * 16 + q] - 1;

  int mrow[8];
#if defined(__AMDGCN__) && __has_builtin(__builtin_amdgcn_wmma_i32_16x16x64_iu8)
  wmma_i32x8 A0, A1, B0, B1;
#pragma unroll
  for (int v = 0; v < 8; ++v) {
    const int ka = ((v & 3) << 2) + (hi << 4) + ((v >= 4) ? 32 : 0);
    const int kb = ((v & 3) << 2) + (hi << 4) + ((v >= 4) ? 32 : 0);
    u32 a0 = 0, a1 = 0, b0 = 0, b1 = 0;
#pragma unroll
    for (int b = 0; b < 4; ++b) {
      if (la  == ka + b)      a0 |= 1u << (8 * b);
      if (la  == ka + b + 64) a1 |= 1u << (8 * b);
      if (lbv == kb + b)      b0 |= 1u << (8 * b);
      if (lbv == kb + b + 64) b1 |= 1u << (8 * b);
    }
    A0[v] = (int)a0; A1[v] = (int)a1; B0[v] = (int)b0; B1[v] = (int)b1;
  }
  wmma_i32x8 acc = (wmma_i32x8)0;
  acc = __builtin_amdgcn_wmma_i32_16x16x64_iu8(false, A0, false, B0, acc, false, false);
  acc = __builtin_amdgcn_wmma_i32_16x16x64_iu8(false, A1, false, B1, acc, false, false);
#pragma unroll
  for (int r = 0; r < 8; ++r) mrow[r] = acc[r];
#else
#pragma unroll
  for (int r = 0; r < 8; ++r) {
    int lr = labn[ti * 16 + r + hi * 8];
    mrow[r] = (lr > 0) && (lr == lbv + 1);
  }
#endif

  const int jg = tj * 16 + q;
  const float jx1 = bx[jg * 4 + 0], jy1 = bx[jg * 4 + 1];
  const float jx2 = bx[jg * 4 + 2], jy2 = bx[jg * 4 + 3];
  const float areaJ = fmaxf(jx2 - jx1, 0.f) * fmaxf(jy2 - jy1, 0.f);
#pragma unroll
  for (int r = 0; r < 8; ++r) {
    const int ig = ti * 16 + r + hi * 8;
    const float ix1 = bx[ig * 4 + 0], iy1 = bx[ig * 4 + 1];
    const float ix2 = bx[ig * 4 + 2], iy2 = bx[ig * 4 + 3];
    const float areaI = fmaxf(ix2 - ix1, 0.f) * fmaxf(iy2 - iy1, 0.f);
    const float iw = fmaxf(fminf(ix2, jx2) - fmaxf(ix1, jx1), 0.f);
    const float ih = fmaxf(fminf(iy2, jy2) - fmaxf(iy1, jy1), 0.f);
    const float inter = iw * ih;
    const float iou = inter / (areaI + areaJ - inter + 1e-9f);
    const bool sup = (mrow[r] != 0) && (iou > IOU_TH) && (jg > ig);
    const u32 bal = ballot32(sup);
    if (l == 2 * r)     mbase[(size_t)(ti * 16 + r) * 64 + tj]     = (u16)(bal & 0xFFFFu);
    if (l == 2 * r + 1) mbase[(size_t)(ti * 16 + r + 8) * 64 + tj] = (u16)(bal >> 16);
  }
}

__global__ void __launch_bounds__(32) lof_scan_out_kernel(
    const u16* __restrict__ mask, const u32* __restrict__ vbits,
    const float* __restrict__ ksc, const float* __restrict__ kbx,
    const int* __restrict__ klb, const int* __restrict__ ktg,
    float* __restrict__ out)
{
  const int n = blockIdx.x;
  const int l = threadIdx.x;
  u32 removed = ~vbits[n * 32 + l];
  u32 keepw = 0;
  const u16* mb = mask + (size_t)n * KCAND * 64;
  for (int i = 0; i < KCAND; ++i) {
    const int w = i >> 5;
    const u32 rw = (u32)__shfl((int)removed, w, 32);
    if (((rw >> (i & 31)) & 1u) == 0u) {
      if (l == w) keepw |= 1u << (i & 31);
      const u16* row = mb + (size_t)i * 64;
      removed |= (u32)row[2 * l] | ((u32)row[2 * l + 1] << 16);
    }
  }
  const int cnt = __popc(keepw);
  int ex = 0, total = 0;
  for (int j = 0; j < 32; ++j) {
    const int cj = __shfl(cnt, j, 32);
    if (j < l) ex += cj;
    total += cj;
  }
  const float* sc = ksc + n * KCAND;
  const float* bx = kbx + (size_t)n * KCAND * 4;
  const int* lb = klb + n * KCAND;
  const int* tg = ktg + n * KCAND;
  float* oB = out + (size_t)n * OUTK * 4;
  float* oS = out + (size_t)NIMG * OUTK * 4 + n * OUTK;
  float* oL = out + (size_t)NIMG * OUTK * 5 + n * OUTK;
  float* oT = out + (size_t)NIMG * OUTK * 6 + n * OUTK;
  float* oV = out + (size_t)NIMG * OUTK * 7 + n * OUTK;
  for (int b = 0; b < 32; ++b) {
    if (keepw & (1u << b)) {
      const int rank = ex + __popc(keepw & ((1u << b) - 1u));
      if (rank < OUTK) {
        const int i = 32 * l + b;
        oB[rank * 4 + 0] = bx[i * 4 + 0]; oB[rank * 4 + 1] = bx[i * 4 + 1];
        oB[rank * 4 + 2] = bx[i * 4 + 2]; oB[rank * 4 + 3] = bx[i * 4 + 3];
        oS[rank] = sc[i];
        oL[rank] = (float)lb[i];
        oT[rank] = (float)tg[i];
        oV[rank] = 1.0f;
      }
    }
  }
  const int kc = total < OUTK ? total : OUTK;
  for (int s = l; s < OUTK; s += 32) {
    if (s >= kc) {
      oB[s * 4 + 0] = 0.f; oB[s * 4 + 1] = 0.f; oB[s * 4 + 2] = 0.f; oB[s * 4 + 3] = 0.f;
      oS[s] = 0.f; oL[s] = 0.f; oT[s] = 0.f; oV[s] = 0.f;
    }
  }
}

extern "C" void kernel_launch(void* const* d_in, const int* in_sizes, int n_in,
                              void* d_out, int out_size, void* d_ws, size_t ws_size,
                              hipStream_t stream)
{
  (void)in_sizes; (void)n_in; (void)out_size; (void)ws_size;
  const float* locations  = (const float*)d_in[0];
  const float* box_cls    = (const float*)d_in[1];
  const float* box_reg    = (const float*)d_in[2];
  const float* centerness = (const float*)d_in[3];
  const float* lof_tag    = (const float*)d_in[4];
  const int*   image_sz   = (const int*)d_in[5];
  float* out = (float*)d_out;

  char* ws = (char*)d_ws;
  size_t off = 0;
  auto take = [&](size_t bytes) -> char* {
    char* p = ws + off;
    off += (bytes + 255) & ~(size_t)255;
    return p;
  };
  float* w_scores = (float*)take((size_t)NIMG * PPIX * sizeof(float));
  int*   w_cls    = (int*)  take((size_t)NIMG * PPIX * sizeof(int));
  float* w_ksc    = (float*)take((size_t)NIMG * KCAND * sizeof(float));
  float* w_kbx    = (float*)take((size_t)NIMG * KCAND * 4 * sizeof(float));
  int*   w_klb    = (int*)  take((size_t)NIMG * KCAND * sizeof(int));
  int*   w_ktg    = (int*)  take((size_t)NIMG * KCAND * sizeof(int));
  u32*   w_vb     = (u32*)  take((size_t)NIMG * 32 * sizeof(u32));
  u16*   w_mask   = (u16*)  take((size_t)NIMG * KCAND * 64 * sizeof(u16));

  lof_score_kernel<<<dim3(PPIX / TILE, NIMG), TILE, 0, stream>>>(
      box_cls, centerness, w_scores, w_cls);
  lof_topk_kernel<<<NIMG, 1024, 0, stream>>>(
      w_scores, w_cls, locations, box_reg, lof_tag, image_sz,
      w_ksc, w_kbx, w_klb, w_ktg, w_vb);
  lof_nms_mask_kernel<<<dim3(KCAND / 16, KCAND / 16, NIMG), 32, 0, stream>>>(
      w_kbx, w_klb, w_mask);
  lof_scan_out_kernel<<<NIMG, 32, 0, stream>>>(
      w_mask, w_vb, w_ksc, w_kbx, w_klb, w_ktg, out);
}
